// AttentionHead_21706764714468
// MI455X (gfx1250) — matrix operand order repaired and checked by reading, not compiled, not run on hardware
//
#include <hip/hip_runtime.h>
#include <hip/hip_bf16.h>
#include <stdint.h>

#define BATCH 512
#define SEQ   256
#define EMB   384
#define HD    64

// Padded LDS leading dims (bf16 elements) to avoid bank conflicts
#define K_LD  72     // K tile rows: 64 data + 8 pad
#define V_LD  264    // Vt rows: 256 data + 8 pad
#define P_LD  264    // P strip rows: 256 data + 8 pad

#define K_BYTES   (SEQ * K_LD * 2)            // 36864
#define V_BYTES   (HD * V_LD * 2)             // 33792
#define P_STRIP   (16 * P_LD * 2)             // 8448 per wave
#define SMEM_SIZE (K_BYTES + V_BYTES + 8 * P_STRIP)   // 138240 B

typedef __attribute__((ext_vector_type(16))) __bf16 v16bf;
typedef __attribute__((ext_vector_type(8)))  __bf16 v8bf;
typedef __attribute__((ext_vector_type(8)))  float  v8f;

// ---------------------------------------------------------------------------
// CDNA5 async global->LDS copy (ASYNCcnt-tracked), §15.18 opcode 98
// ---------------------------------------------------------------------------
__device__ __forceinline__ void async_load16(const void* lds_dst, const void* gsrc) {
    unsigned lds_off = (unsigned)(uintptr_t)lds_dst;              // low 32 bits = LDS offset
    unsigned long long ga = (unsigned long long)(uintptr_t)gsrc;
    asm volatile("global_load_async_to_lds_b128 %0, %1, off"
                 :: "v"(lds_off), "v"(ga) : "memory");
}
__device__ __forceinline__ void wait_async0() {
    asm volatile("s_wait_asynccnt 0x0" ::: "memory");
}

// ---------------------------------------------------------------------------
// WMMA fragment helpers (v_wmma_f32_16x16x32_bf16 layouts, ISA 7.12.2)
// ---------------------------------------------------------------------------

// A fragment (16x32 MxK), row-major bf16, leading dim ld (elements).
__device__ __forceinline__ v16bf load_a_frag(const __bf16* src, int ld,
                                             int row0, int k0, int half, int l16) {
    const __bf16* p = src + (size_t)(row0 + l16) * ld + k0 + half * 8;
    v8bf lo = *(const v8bf*)(p);
    v8bf hi = *(const v8bf*)(p + 16);
    v16bf a;
#pragma unroll
    for (int i = 0; i < 8; ++i) { a[i] = lo[i]; a[i + 8] = hi[i]; }
    return a;
}

// B fragment (32x16 KxN) from K-contiguous ([N][K]) storage, leading dim ldt.
__device__ __forceinline__ v16bf load_b_frag(const __bf16* srcT, int ldt,
                                             int n0, int k0, int half, int l16) {
    const __bf16* p = srcT + (size_t)(n0 + l16) * ldt + k0 + half * 8;
    v8bf lo = *(const v8bf*)(p);
    v8bf hi = *(const v8bf*)(p + 16);
    v16bf b;
#pragma unroll
    for (int i = 0; i < 8; ++i) { b[i] = lo[i]; b[i + 8] = hi[i]; }
    return b;
}

__device__ __forceinline__ v8f wmma_bf16(v16bf a, v16bf b, v8f c) {
    return __builtin_amdgcn_wmma_f32_16x16x32_bf16(false, a, false, b,
                                                   (short)0, c, false, false);
}

// ---------------------------------------------------------------------------
// Kernel 0: convert + transpose weights: W[E][D] f32 -> Wt[D][E] bf16
// ---------------------------------------------------------------------------
__global__ void convert_w(const float* __restrict__ Wk,
                          const float* __restrict__ Wq,
                          const float* __restrict__ Wv,
                          __bf16* __restrict__ WtQ,
                          __bf16* __restrict__ WtK,
                          __bf16* __restrict__ WtV) {
    int idx = blockIdx.x * blockDim.x + threadIdx.x;   // over [HD][EMB]
    if (idx >= HD * EMB) return;
    int d = idx / EMB;
    int e = idx % EMB;
    WtQ[idx] = (__bf16)Wq[e * HD + d];
    WtK[idx] = (__bf16)Wk[e * HD + d];
    WtV[idx] = (__bf16)Wv[e * HD + d];
}

// ---------------------------------------------------------------------------
// Kernel 1: QKV projection.  C[M=B*S, N=192] = X[M,384] * [Wq|Wk|Wv]
// 8 waves/block, wave owns 16 rows x 12 N-tiles. Writes Q,K row-major bf16
// and V transposed (Vt[B][D][S]) bf16.  HBM-bound on streaming X (read once).
// ---------------------------------------------------------------------------
__global__ __launch_bounds__(256) void qkv_proj(
        const float* __restrict__ X,
        const __bf16* __restrict__ WtQ,
        const __bf16* __restrict__ WtK,
        const __bf16* __restrict__ WtV,
        __bf16* __restrict__ Q,
        __bf16* __restrict__ K,
        __bf16* __restrict__ Vt) {
    const int wave = threadIdx.x >> 5;
    const int lane = threadIdx.x & 31;
    const int half = lane >> 4;
    const int l16  = lane & 15;
    const int m0   = blockIdx.x * 128 + wave * 16;     // 16-row band (never straddles a batch)

    v8f acc[12];
#pragma unroll
    for (int t = 0; t < 12; ++t) acc[t] = (v8f)0.0f;

    for (int ks = 0; ks < 12; ++ks) {
        const int k0 = ks * 32;
        // A fragment: convert X (f32) inline
        const float* px = X + (size_t)(m0 + l16) * EMB + k0 + half * 8;
        const float4 x0 = *(const float4*)(px + 0);
        const float4 x1 = *(const float4*)(px + 4);
        const float4 x2 = *(const float4*)(px + 16);
        const float4 x3 = *(const float4*)(px + 20);
        v16bf a;
        a[0]  = (__bf16)x0.x; a[1]  = (__bf16)x0.y; a[2]  = (__bf16)x0.z; a[3]  = (__bf16)x0.w;
        a[4]  = (__bf16)x1.x; a[5]  = (__bf16)x1.y; a[6]  = (__bf16)x1.z; a[7]  = (__bf16)x1.w;
        a[8]  = (__bf16)x2.x; a[9]  = (__bf16)x2.y; a[10] = (__bf16)x2.z; a[11] = (__bf16)x2.w;
        a[12] = (__bf16)x3.x; a[13] = (__bf16)x3.y; a[14] = (__bf16)x3.z; a[15] = (__bf16)x3.w;

#pragma unroll
        for (int nt = 0; nt < 12; ++nt) {
            const __bf16* wt = (nt < 4) ? WtQ : (nt < 8) ? WtK : WtV;
            v16bf b = load_b_frag(wt, EMB, (nt & 3) * 16, k0, half, l16);
            acc[nt] = wmma_bf16(a, b, acc[nt]);
        }
    }

    // Store Q, K row-major bf16: C layout -> row m0 + r + half*8, col d0 + l16
#pragma unroll
    for (int nt = 0; nt < 8; ++nt) {
        __bf16* dst = (nt < 4) ? Q : K;
        const int d0 = (nt & 3) * 16;
#pragma unroll
        for (int r = 0; r < 8; ++r) {
            const int m = m0 + r + half * 8;
            dst[(size_t)m * HD + d0 + l16] = (__bf16)acc[nt][r];
        }
    }

    // Store V transposed: Vt[b][d][s]; lane owns column d, 8 consecutive s.
    const int bb = m0 / SEQ;
    const int s0 = m0 % SEQ;
#pragma unroll
    for (int nt = 8; nt < 12; ++nt) {
        const int d = (nt - 8) * 16 + l16;
        v8bf pk;
#pragma unroll
        for (int r = 0; r < 8; ++r) pk[r] = (__bf16)acc[nt][r];
        *(v8bf*)(Vt + ((size_t)bb * HD + d) * SEQ + s0 + half * 8) = pk;
    }
}

// ---------------------------------------------------------------------------
// Kernel 2: causal attention.  One block per batch.
// K[b] and Vt[b] async-staged to LDS once per block; each wave processes
// query tiles {w, 15-w} sequentially => uniform 17 key-tiles of score work.
// ---------------------------------------------------------------------------
__global__ __launch_bounds__(256) void attention(
        const __bf16* __restrict__ Q,
        const __bf16* __restrict__ K,
        const __bf16* __restrict__ Vt,
        float* __restrict__ Out) {
    extern __shared__ char smem[];
    __bf16* Klds = (__bf16*)(smem);                         // [SEQ][K_LD]
    __bf16* Vlds = (__bf16*)(smem + K_BYTES);               // [HD][V_LD]

    const int wave = threadIdx.x >> 5;
    const int lane = threadIdx.x & 31;
    const int half = lane >> 4;
    const int l16  = lane & 15;

    __bf16* Pw = (__bf16*)(smem + K_BYTES + V_BYTES + wave * P_STRIP);  // [16][P_LD]

    const int bidx = blockIdx.x;
    const __bf16* Qb  = Q  + (size_t)bidx * SEQ * HD;
    const __bf16* Kb  = K  + (size_t)bidx * SEQ * HD;
    const __bf16* Vtb = Vt + (size_t)bidx * HD * SEQ;

    // ---- async-stage K (256x64) and Vt (64x256) into padded LDS ----
#pragma unroll
    for (int i = 0; i < 8; ++i) {
        const int chunk = i * 256 + threadIdx.x;            // 2048 x 16B chunks each
        const int rk = chunk >> 3, ck = chunk & 7;          // K: 8 chunks per 64-elem row
        async_load16(Klds + rk * K_LD + ck * 8, Kb + rk * HD + ck * 8);
        const int rv = chunk >> 5, cv = chunk & 31;         // Vt: 32 chunks per 256-elem row
        async_load16(Vlds + rv * V_LD + cv * 8, Vtb + rv * SEQ + cv * 8);
    }
    wait_async0();
    __syncthreads();

    const float NEG_INF = -__builtin_inff();

#pragma unroll 1
    for (int pass = 0; pass < 2; ++pass) {
        const int qtile  = pass ? (15 - wave) : wave;       // {w, 15-w}
        const int q0     = qtile * 16;
        const int ntiles = qtile + 1;                       // causal key tiles

        // Q A-fragments for the two 32-wide K-steps over D=64
        const v16bf aq0 = load_a_frag(Qb, HD, q0, 0,  half, l16);
        const v16bf aq1 = load_a_frag(Qb, HD, q0, 32, half, l16);

        // ---- scores (causally skipped, wave-uniform branches) ----
        v8f sacc[16];
#pragma unroll
        for (int t = 0; t < 16; ++t) sacc[t] = (v8f)0.0f;

#pragma unroll
        for (int kt = 0; kt < 16; ++kt) {
            if (kt < ntiles) {
                v16bf bk0 = load_b_frag(Klds, K_LD, kt * 16, 0,  half, l16);
                v16bf bk1 = load_b_frag(Klds, K_LD, kt * 16, 32, half, l16);
                sacc[kt] = wmma_bf16(aq0, bk0, sacc[kt]);
                sacc[kt] = wmma_bf16(aq1, bk1, sacc[kt]);
            }
        }

        // ---- scale + causal mask + row max (rows live in 16-lane halves) ----
        float rowmax[8];
#pragma unroll
        for (int r = 0; r < 8; ++r) {
            const int rloc = r + half * 8;
            float mx = NEG_INF;
#pragma unroll
            for (int kt = 0; kt < 16; ++kt) {
                if (kt < ntiles) {
                    float s = sacc[kt][r] * 0.125f;         // 1/sqrt(64)
                    const bool valid = (kt < qtile) || (l16 <= rloc);
                    s = valid ? s : NEG_INF;
                    sacc[kt][r] = s;
                    mx = fmaxf(mx, s);
                }
            }
#pragma unroll
            for (int off = 1; off < 16; off <<= 1)
                mx = fmaxf(mx, __shfl_xor(mx, off, 32));
            rowmax[r] = mx;
        }

        // ---- exp + row sum ----
        float rsum[8];
#pragma unroll
        for (int r = 0; r < 8; ++r) {
            float sm = 0.0f;
#pragma unroll
            for (int kt = 0; kt < 16; ++kt) {
                if (kt < ntiles) {
                    float e = __expf(sacc[kt][r] - rowmax[r]);
                    sacc[kt][r] = e;
                    sm += e;
                }
            }
#pragma unroll
            for (int off = 1; off < 16; off <<= 1)
                sm += __shfl_xor(sm, off, 32);
            rsum[r] = sm;
        }

        // ---- normalize, write P (bf16) to this wave's private LDS strip ----
#pragma unroll
        for (int kt = 0; kt < 16; ++kt) {
            if (kt < ntiles) {
#pragma unroll
                for (int r = 0; r < 8; ++r)
                    Pw[(r + half * 8) * P_LD + kt * 16 + l16] =
                        (__bf16)(sacc[kt][r] / rsum[r]);
            } else {
#pragma unroll
                for (int r = 0; r < 8; ++r)
                    Pw[(r + half * 8) * P_LD + kt * 16 + l16] = (__bf16)0.0f;
            }
        }

        // ---- out = P @ V  (32-wide key steps, B from LDS Vt) ----
        v8f oacc[4];
#pragma unroll
        for (int t = 0; t < 4; ++t) oacc[t] = (v8f)0.0f;

        const int ksteps = (ntiles + 1) >> 1;
#pragma unroll
        for (int ksp = 0; ksp < 8; ++ksp) {
            if (ksp < ksteps) {
                v16bf ap = load_a_frag(Pw, P_LD, 0, ksp * 32, half, l16);
#pragma unroll
                for (int nt = 0; nt < 4; ++nt) {
                    v16bf bv = load_b_frag(Vlds, V_LD, nt * 16, ksp * 32, half, l16);
                    oacc[nt] = wmma_bf16(ap, bv, oacc[nt]);
                }
            }
        }

        // ---- store f32 output ----
        float* Ob = Out + ((size_t)bidx * SEQ + q0) * HD;
#pragma unroll
        for (int nt = 0; nt < 4; ++nt) {
#pragma unroll
            for (int r = 0; r < 8; ++r)
                Ob[(size_t)(r + half * 8) * HD + nt * 16 + l16] = oacc[nt][r];
        }
    }
}

// ---------------------------------------------------------------------------
// Launcher
// ---------------------------------------------------------------------------
extern "C" void kernel_launch(void* const* d_in, const int* in_sizes, int n_in,
                              void* d_out, int out_size, void* d_ws, size_t ws_size,
                              hipStream_t stream) {
    const float* X  = (const float*)d_in[0];
    const float* Wk = (const float*)d_in[1];
    const float* Wq = (const float*)d_in[2];
    const float* Wv = (const float*)d_in[3];
    float* Out = (float*)d_out;

    // Workspace layout (bytes)
    char* ws = (char*)d_ws;
    const size_t wt_bytes  = (size_t)HD * EMB * sizeof(__bf16);          // 48 KB each
    const size_t qkv_bytes = (size_t)BATCH * SEQ * HD * sizeof(__bf16);  // 16 MB each
    __bf16* WtQ = (__bf16*)(ws);
    __bf16* WtK = (__bf16*)(ws + wt_bytes);
    __bf16* WtV = (__bf16*)(ws + 2 * wt_bytes);
    __bf16* Qb  = (__bf16*)(ws + 3 * wt_bytes);
    __bf16* Kb  = (__bf16*)(ws + 3 * wt_bytes + qkv_bytes);
    __bf16* Vt  = (__bf16*)(ws + 3 * wt_bytes + 2 * qkv_bytes);

    // 0: weight convert+transpose
    convert_w<<<(HD * EMB + 255) / 256, 256, 0, stream>>>(Wk, Wq, Wv, WtQ, WtK, WtV);

    // 1: QKV projection (M = B*S = 131072, 128 rows per block)
    qkv_proj<<<(BATCH * SEQ) / 128, 256, 0, stream>>>(X, WtQ, WtK, WtV, Qb, Kb, Vt);

    // 2: causal attention (one block per batch, 135 KB dynamic LDS)
    attention<<<BATCH, 256, SMEM_SIZE, stream>>>(Qb, Kb, Vt, Out);
}
